// Net_39041252721137
// MI455X (gfx1250) — matrix operand order repaired and checked by reading, not compiled, not run on hardware
//
#include <hip/hip_runtime.h>

typedef __attribute__((ext_vector_type(16))) _Float16 v16h;
typedef __attribute__((ext_vector_type(8)))  float    v8f;

#define B_ROWS 16384
#define D_IN   784
#define KP     800      // K padded to multiple of 32
#define N2     8192
#define NOUT   10

#define BM 128
#define BN 128
#define BK 32
#define NSTEP (KP / BK)        // 25
#define LSTR  40               // LDS row stride in halves (pad, keeps 16B align)
#define A_BYTES (BM * LSTR * 2)      // 10240
#define B_BYTES (BN * LSTR * 2)      // 10240
#define STAGE   (A_BYTES + B_BYTES)  // 20480
#define NBUF    3                    // triple-buffered stages

struct __align__(16) U4 { unsigned x, y, z, w; };

// ---- CDNA5 async memory->LDS copy (ASYNCcnt-tracked, no VGPR staging) ----
__device__ __forceinline__ void async_cp16(unsigned lds_addr, const void* gptr) {
    asm volatile("global_load_async_to_lds_b128 %0, %1, off"
                 :: "v"(lds_addr), "v"((unsigned long long)(uintptr_t)gptr)
                 : "memory");
}
__device__ __forceinline__ void wait_async0() {
#if __has_builtin(__builtin_amdgcn_s_wait_asynccnt)
    __builtin_amdgcn_s_wait_asynccnt(0);
#else
    asm volatile("s_wait_asynccnt 0" ::: "memory");
#endif
}
// async ops complete in order: <=4 outstanding means the oldest stage (4 ops) landed
__device__ __forceinline__ void wait_async_le4() {
#if __has_builtin(__builtin_amdgcn_s_wait_asynccnt)
    __builtin_amdgcn_s_wait_asynccnt(4);
#else
    asm volatile("s_wait_asynccnt 4" ::: "memory");
#endif
}

// ---------------- utility ----------------
__global__ void k_zero(unsigned* p, int n) {
    int i = blockIdx.x * blockDim.x + threadIdx.x;
    if (i < n) p[i] = 0u;
}

// ---------------- radix select of rank-j |s| ----------------
__global__ void k_hist_hi(const float* __restrict__ s, int n, unsigned* __restrict__ hist) {
    int i = blockIdx.x * blockDim.x + threadIdx.x;
    if (i >= n) return;
    unsigned bits = __float_as_uint(s[i]) & 0x7fffffffu;
    atomicAdd(&hist[bits >> 16], 1u);
}

__global__ void k_hist_lo(const float* __restrict__ s, int n,
                          unsigned* __restrict__ hist,
                          const unsigned long long* __restrict__ state) {
    unsigned hb = (unsigned)state[0];
    int i = blockIdx.x * blockDim.x + threadIdx.x;
    if (i >= n) return;
    unsigned bits = __float_as_uint(s[i]) & 0x7fffffffu;
    if ((bits >> 16) == hb) atomicAdd(&hist[bits & 0xffffu], 1u);
}

__global__ void k_select(const unsigned* __restrict__ hist, unsigned long long target0,
                         unsigned long long* state, int phase) {
    if (threadIdx.x != 0 || blockIdx.x != 0) return;
    unsigned long long target = (phase == 0) ? target0 : state[1];
    unsigned long long cum = 0;
    unsigned b = 65535u;
    for (unsigned i = 0; i < 65536u; ++i) {
        unsigned c = hist[i];
        if (cum + c > target) { b = i; break; }
        cum += c;
    }
    if (phase == 0) { state[0] = b; state[1] = target - cum; }
    else            { state[2] = ((state[0] & 0xffffull) << 16) | b; }
}

// ---------------- pack to f16 (masked, K-padded) ----------------
__global__ void k_pack_x(const float* __restrict__ x, _Float16* __restrict__ xf) {
    int i = blockIdx.x * blockDim.x + threadIdx.x;
    if (i >= B_ROWS * KP) return;
    int r = i / KP, k = i - r * KP;
    xf[i] = (k < D_IN) ? (_Float16)x[r * D_IN + k] : (_Float16)0.0f;
}

__global__ void k_pack_w1(const float* __restrict__ w, const float* __restrict__ s,
                          const unsigned long long* __restrict__ state,
                          _Float16* __restrict__ wf) {
    unsigned tb = (unsigned)state[2];
    int i = blockIdx.x * blockDim.x + threadIdx.x;
    if (i >= N2 * KP) return;
    int r = i / KP, k = i - r * KP;
    _Float16 v = (_Float16)0.0f;
    if (k < D_IN) {
        unsigned sb = __float_as_uint(s[r * D_IN + k]) & 0x7fffffffu;
        if (sb >= tb) v = (_Float16)w[r * D_IN + k];
    }
    wf[i] = v;
}

__global__ void k_pack_w2(const float* __restrict__ w, const float* __restrict__ s,
                          const unsigned long long* __restrict__ state,
                          float* __restrict__ wm) {
    unsigned tb = (unsigned)state[2];
    int i = blockIdx.x * blockDim.x + threadIdx.x;
    if (i >= NOUT * N2) return;
    unsigned sb = __float_as_uint(s[i]) & 0x7fffffffu;
    wm[i] = (sb >= tb) ? w[i] : 0.0f;
}

// ---- fused masked-GEMM1 (async triple-buffered) + relu + partial GEMM2 ----
__global__ __launch_bounds__(256)
void k_gemm_fused(const _Float16* __restrict__ X, const _Float16* __restrict__ W1,
                  const float* __restrict__ W2m, float* __restrict__ logits) {
    __shared__ __align__(16) char smem[NBUF * STAGE];  // 61440 B; Hs aliases after k-loop

    const int tid  = threadIdx.x;
    const int lane = tid & 31;
    const int wid  = tid >> 5;          // 0..7
    const int wm   = wid & 3;           // 4 M sub-tiles of 32
    const int wn   = wid >> 2;          // 2 N sub-tiles of 64
    const int m0   = blockIdx.x * BM;
    const int n0   = blockIdx.y * BN;

    // per-thread copy assignments: A 512 chunks of 16B, B 512 chunks, 2+2 each
    const int ra0 = tid >> 2,          ca0 = (tid & 3) * 8;
    const int ra1 = (tid + 256) >> 2,  ca1 = ((tid + 256) & 3) * 8;
    const _Float16* gA0 = X  + (size_t)(m0 + ra0) * KP + ca0;
    const _Float16* gA1 = X  + (size_t)(m0 + ra1) * KP + ca1;
    const _Float16* gB0 = W1 + (size_t)(n0 + ra0) * KP + ca0;
    const _Float16* gB1 = W1 + (size_t)(n0 + ra1) * KP + ca1;
    const unsigned smem_base = (unsigned)(uintptr_t)smem;
    const unsigned lA0 = (unsigned)((ra0 * LSTR + ca0) * 2);
    const unsigned lA1 = (unsigned)((ra1 * LSTR + ca1) * 2);
    const unsigned lB0 = (unsigned)(A_BYTES + (ra0 * LSTR + ca0) * 2);
    const unsigned lB1 = (unsigned)(A_BYTES + (ra1 * LSTR + ca1) * 2);

    auto issue_stage = [&](int k0, int buf) {
        unsigned sb = smem_base + (unsigned)(buf * STAGE);
        async_cp16(sb + lA0, gA0 + k0);
        async_cp16(sb + lA1, gA1 + k0);
        async_cp16(sb + lB0, gB0 + k0);
        async_cp16(sb + lB1, gB1 + k0);
    };

    v8f acc[2][4];
    #pragma unroll
    for (int i = 0; i < 2; ++i)
        #pragma unroll
        for (int j = 0; j < 4; ++j)
            #pragma unroll
            for (int e = 0; e < 8; ++e) acc[i][j][e] = 0.0f;

    const int rlow = lane & 15;
    const int rsel = (lane >> 4) * 8;    // A: K 0..7 / 8..15 then +16
    const int ksel = (lane >> 4) * 16;   // unused: B loads use rsel, same K order as A

    // prologue: two stages in flight (8 outstanding async ops per wave)
    issue_stage(0, 0);
    issue_stage(BK, 1);
    int cur = 0, pre = 2;                // pre = buffer for stage step+2
    for (int step = 0; step < NSTEP; ++step) {
        if (step == NSTEP - 1) wait_async0();  // only last stage outstanding
        else                   wait_async_le4(); // oldest stage (4 ops) landed
        __syncthreads();        // all waves' copies landed; reads of `pre` (step-1) done
        if (step + 2 < NSTEP) issue_stage((step + 2) * BK, pre);

        const _Float16* Asp = (const _Float16*)(smem + cur * STAGE);
        const _Float16* Bsp = (const _Float16*)(smem + cur * STAGE + A_BYTES);
        union F { v16h v; U4 u[2]; } a[2], b[4];
        #pragma unroll
        for (int t = 0; t < 2; ++t) {
            int r = wm * 32 + t * 16 + rlow;
            a[t].u[0] = *(const U4*)&Asp[r * LSTR + rsel];
            a[t].u[1] = *(const U4*)&Asp[r * LSTR + 16 + rsel];
        }
        #pragma unroll
        for (int j = 0; j < 4; ++j) {
            int c = wn * 64 + j * 16 + rlow;
            b[j].u[0] = *(const U4*)&Bsp[c * LSTR + rsel];
            b[j].u[1] = *(const U4*)&Bsp[c * LSTR + 16 + rsel];
        }
        #pragma unroll
        for (int i = 0; i < 2; ++i)
            #pragma unroll
            for (int j = 0; j < 4; ++j)
                acc[i][j] = __builtin_amdgcn_wmma_f32_16x16x32_f16(
                    false, a[i].v, false, b[j].v, (short)0, acc[i][j], false, false);

        cur = (cur == NBUF - 1) ? 0 : cur + 1;
        pre = (pre == NBUF - 1) ? 0 : pre + 1;
    }

    // ---- epilogue: relu h-tile into LDS (aliases staging buffers), f16 ----
    __syncthreads();
    _Float16 (*Hs)[BN] = (_Float16 (*)[BN])smem;    // 128x128 f16 = 32768 B
    #pragma unroll
    for (int i = 0; i < 2; ++i)
        #pragma unroll
        for (int j = 0; j < 4; ++j) {
            int rbase = wm * 32 + i * 16 + ((lane >> 4) ? 8 : 0);
            int c     = wn * 64 + j * 16 + rlow;
            #pragma unroll
            for (int e = 0; e < 8; ++e) {
                float v = acc[i][j][e];
                Hs[rbase + e][c] = (_Float16)(v > 0.0f ? v : 0.0f);
            }
        }
    __syncthreads();

    // partial layer-2: 128 rows x 10 outputs over this block's 128 columns
    for (int p = tid; p < BM * NOUT; p += 256) {
        int m = p / NOUT, o = p - m * NOUT;
        const float* wrow = &W2m[(size_t)o * N2 + n0];
        float sum = 0.0f;
        #pragma unroll 8
        for (int c = 0; c < BN; ++c) sum += (float)Hs[m][c] * wrow[c];
        atomicAdd(&logits[(size_t)(m0 + m) * NOUT + o], sum);
    }
}

// ---------------- log_softmax over 10 logits per row ----------------
__global__ void k_logsoftmax(const float* __restrict__ logits, float* __restrict__ out) {
    int r = blockIdx.x * blockDim.x + threadIdx.x;
    if (r >= B_ROWS) return;
    float v[NOUT];
    float mx = -__builtin_inff();
    #pragma unroll
    for (int o = 0; o < NOUT; ++o) { v[o] = logits[r * NOUT + o]; mx = fmaxf(mx, v[o]); }
    float s = 0.0f;
    #pragma unroll
    for (int o = 0; o < NOUT; ++o) s += __expf(v[o] - mx);
    float l = __logf(s);
    #pragma unroll
    for (int o = 0; o < NOUT; ++o) out[r * NOUT + o] = v[o] - mx - l;
}

extern "C" void kernel_launch(void* const* d_in, const int* in_sizes, int n_in,
                              void* d_out, int out_size, void* d_ws, size_t ws_size,
                              hipStream_t stream) {
    (void)in_sizes; (void)n_in; (void)out_size; (void)ws_size;
    const float* x  = (const float*)d_in[0];
    const float* w1 = (const float*)d_in[1];
    const float* s1 = (const float*)d_in[2];
    const float* w2 = (const float*)d_in[3];
    const float* s2 = (const float*)d_in[4];

    char* ws = (char*)d_ws;
    size_t off = 0;
    auto take = [&](size_t bytes) -> char* {
        char* p = ws + off;
        off = (off + bytes + 255) & ~(size_t)255;
        return p;
    };
    unsigned*            hist   = (unsigned*)take(65536 * sizeof(unsigned));
    unsigned long long*  st1    = (unsigned long long*)take(4 * sizeof(unsigned long long));
    unsigned long long*  st2    = (unsigned long long*)take(4 * sizeof(unsigned long long));
    _Float16*            Xf     = (_Float16*)take((size_t)B_ROWS * KP * sizeof(_Float16));
    _Float16*            W1f    = (_Float16*)take((size_t)N2 * KP * sizeof(_Float16));
    float*               W2m    = (float*)take((size_t)NOUT * N2 * sizeof(float));
    float*               logits = (float*)take((size_t)B_ROWS * NOUT * sizeof(float));

    const int T = 256;
    const int n1 = N2 * D_IN;                 // 6,422,528
    const unsigned long long j1 = (unsigned long long)(n1 / 2);
    const int n2 = NOUT * N2;                 // 81,920
    const unsigned long long j2 = (unsigned long long)(n2 / 2);

    // --- threshold for s1 (rank n/2 of |s1|, two-pass radix select) ---
    k_zero   <<<(65536 + T - 1) / T, T, 0, stream>>>(hist, 65536);
    k_hist_hi<<<(n1 + T - 1) / T,   T, 0, stream>>>(s1, n1, hist);
    k_select <<<1, 1, 0, stream>>>(hist, j1, st1, 0);
    k_zero   <<<(65536 + T - 1) / T, T, 0, stream>>>(hist, 65536);
    k_hist_lo<<<(n1 + T - 1) / T,   T, 0, stream>>>(s1, n1, hist, st1);
    k_select <<<1, 1, 0, stream>>>(hist, 0ull, st1, 1);

    // --- threshold for s2 ---
    k_zero   <<<(65536 + T - 1) / T, T, 0, stream>>>(hist, 65536);
    k_hist_hi<<<(n2 + T - 1) / T,   T, 0, stream>>>(s2, n2, hist);
    k_select <<<1, 1, 0, stream>>>(hist, j2, st2, 0);
    k_zero   <<<(65536 + T - 1) / T, T, 0, stream>>>(hist, 65536);
    k_hist_lo<<<(n2 + T - 1) / T,   T, 0, stream>>>(s2, n2, hist, st2);
    k_select <<<1, 1, 0, stream>>>(hist, 0ull, st2, 1);

    // --- pack operands (f16, K padded 784 -> 800, mask applied to w1/w2) ---
    k_pack_x <<<((B_ROWS * KP) + T - 1) / T, T, 0, stream>>>(x, Xf);
    k_pack_w1<<<((N2 * KP) + T - 1) / T,     T, 0, stream>>>(w1, s1, st1, W1f);
    k_pack_w2<<<((NOUT * N2) + T - 1) / T,   T, 0, stream>>>(w2, s2, st2, W2m);

    // --- zero logits accumulator, fused GEMMs, log-softmax ---
    k_zero<<<((B_ROWS * NOUT) + T - 1) / T, T, 0, stream>>>((unsigned*)logits, B_ROWS * NOUT);
    dim3 grid(B_ROWS / BM, N2 / BN);          // 128 x 64 blocks
    k_gemm_fused<<<grid, 256, 0, stream>>>(Xf, W1f, W2m, logits);
    k_logsoftmax<<<(B_ROWS + T - 1) / T, T, 0, stream>>>(logits, (float*)d_out);
}
